// SwinWindowAttention_75823352643992
// MI455X (gfx1250) — matrix operand order repaired and checked by reading, not compiled, not run on hardware
//
#include <hip/hip_runtime.h>

typedef __attribute__((ext_vector_type(16))) _Float16 v16h;
typedef __attribute__((ext_vector_type(8)))  float    v8f;

#define NUM_H 4
#define CSZ   128
#define NW    49
#define NWIN  64

// ---------------- workspace layout (bytes) ----------------
// wqkv_frag : [nt=24][kt=4][lane=32][e=16] f16   @ 0       (98304 B)
// wproj_frag: [nt= 8][kt=4][lane=32][e=16] f16   @ 98304   (32768 B)
// biasTbl   : [4][49][49] f32                    @ 131072  (38416 B)

__global__ void swin_prep_kernel(const float* __restrict__ qkv_w,
                                 const float* __restrict__ proj_w,
                                 const float* __restrict__ rel_table,
                                 _Float16* __restrict__ wqkv,
                                 _Float16* __restrict__ wproj,
                                 float* __restrict__ biasTbl) {
  const int total = 49152 + 16384 + 9604;
  for (int f = blockIdx.x * blockDim.x + threadIdx.x; f < total;
       f += gridDim.x * blockDim.x) {
    if (f < 49152) {                 // qkv weight -> B-fragment order (f16)
      int e = f & 15, lane = (f >> 4) & 31, kt = (f >> 9) & 3, nt = f >> 11;
      int n = nt * 16 + (lane & 15);
      int k = kt * 32 + ((lane >> 4) << 4) + e;   // B: lane-hi selects K half (16 contiguous)
      wqkv[f] = (_Float16)qkv_w[k * 384 + n];
    } else if (f < 65536) {          // proj weight -> B-fragment order (f16)
      int g = f - 49152;
      int e = g & 15, lane = (g >> 4) & 31, kt = (g >> 9) & 3, nt = g >> 11;
      int n = nt * 16 + (lane & 15);
      int k = kt * 32 + ((lane >> 4) << 4) + e;
      wproj[g] = (_Float16)proj_w[k * 128 + n];
    } else {                         // relative position bias table [h][i][j]
      int g = f - 65536;
      int h = g / 2401, rem = g % 2401;
      int i = rem / 49, j = rem % 49;
      int dy = i / 7 - j / 7 + 6, dx = i % 7 - j % 7 + 6;
      biasTbl[g] = rel_table[(dy * 13 + dx) * NUM_H + h];
    }
  }
}

// A-matrix fragment (16x32 f16): lane holds row m=lane&15; K = (lane>>4)*16 + 0..15
// contiguous.  Two 16-B LDS loads.
__device__ __forceinline__ v16h fragA(const _Float16* base, int row0, int stride,
                                      int kbase, int lane) {
  const _Float16* p = base + (row0 + (lane & 15)) * stride + kbase + ((lane >> 4) << 4);
  union { v16h v; float4 f[2]; } u;
  u.f[0] = *(const float4*)p;
  u.f[1] = *(const float4*)(p + 8);
  return u.v;
}

// B-matrix fragment (32x16 f16): lane holds col n=lane&15; K = (lane>>4)*16 + 0..15
// contiguous.  Source stored as [N][K] row-major (i.e. B-transposed).
__device__ __forceinline__ v16h fragB(const _Float16* base, int col0, int stride,
                                      int kbase, int lane) {
  const _Float16* p = base + (col0 + (lane & 15)) * stride + kbase + ((lane >> 4) << 4);
  union { v16h v; float4 f[2]; } u;
  u.f[0] = ((const float4*)p)[0];
  u.f[1] = ((const float4*)p)[1];
  return u.v;
}

__device__ __forceinline__ v8f wmma_f16(v16h a, v16h b, v8f c) {
  return __builtin_amdgcn_wmma_f32_16x16x32_f16(false, a, false, b, (short)0, c,
                                                false, false);
}

__global__ __launch_bounds__(256)
void swin_attn_kernel(const float* __restrict__ x_in, const float* __restrict__ mask,
                      const float* __restrict__ qkv_b, const float* __restrict__ proj_b,
                      const _Float16* __restrict__ wqkv, const _Float16* __restrict__ wproj,
                      const float* __restrict__ biasTbl, float* __restrict__ out) {
  // 64 KB LDS, phase-overlapped:
  //   [    0, 8192) qs  [4][64][32]   -> later as_ (low half)
  //   [ 8192,16384) ks  [4][64][32]   -> later as_ (high half)
  //   [16384,24576) vts [4][32][64]
  //   [24576,32768) xs  [64][128]     -> later ys [64][128]
  __shared__ __align__(16) _Float16 smem[32768];
  _Float16* qs  = smem;
  _Float16* ks  = smem + 8192;
  _Float16* vts = smem + 16384;
  _Float16* xs  = smem + 24576;
  _Float16* as_ = smem;          // [4][64][64] attn probs (after barrier)
  _Float16* ys  = smem + 24576;  // [64][128]

  const int b = blockIdx.x;
  const int win = b & (NWIN - 1);
  const int tid = threadIdx.x;
  const int w = tid >> 5, lane = tid & 31;
  const int laneN = lane & 15, laneHi = lane >> 4;

  // ---- 1. stage x (pad M 49->64 with zeros), f32 -> f16 ----
  for (int idx = tid; idx < 64 * 128; idx += 256) {
    int m = idx >> 7, c = idx & 127;
    float v = (m < NW) ? x_in[((size_t)b * NW + m) * CSZ + c] : 0.f;
    xs[idx] = (_Float16)v;
  }
  __syncthreads();

  // ---- 2. QKV GEMM: [64x128] @ [128x384] ; 96 C-tiles, 12 per wave ----
  const float scale = 0.17677669529663687f;  // 32^-0.5
  for (int i = 0; i < 12; ++i) {
    int t = w + 8 * i;                // 0..95
    int mt = t & 3, nt = t >> 2;
    v8f acc = {};
    for (int kt = 0; kt < 4; ++kt) {
      v16h a  = fragA(xs, mt * 16, 128, kt * 32, lane);
      v16h bb = *(const v16h*)(wqkv + (((size_t)nt * 4 + kt) * 32 + lane) * 16);
      acc = wmma_f16(a, bb, acc);
    }
    int n = nt * 16 + laneN;
    float bias = qkv_b[n];
    for (int r = 0; r < 8; ++r) {
      int m = mt * 16 + r + laneHi * 8;
      float val = acc[r] + bias;
      if (n < 128) {                                    // q (pre-scaled)
        int h = n >> 5, d = n & 31;
        qs[h * 2048 + m * 32 + d] = (_Float16)(val * scale);
      } else if (n < 256) {                             // k  ([N][K] = row major)
        int n2 = n - 128; int h = n2 >> 5, d = n2 & 31;
        ks[h * 2048 + m * 32 + d] = (_Float16)val;
      } else {                                          // v transposed: [d][m]
        int n2 = n - 256; int h = n2 >> 5, d = n2 & 31;
        vts[h * 2048 + d * 64 + m] = (_Float16)val;
      }
    }
  }
  __syncthreads();

  // ---- 3. scores q.kT + bias + mask + softmax; 16 strips, 2/wave, in regs ----
  v8f p[2][4];
  for (int i = 0; i < 2; ++i) {
    int s = w + 8 * i;                // 0..15
    int h = s >> 2, mt = s & 3;
    for (int nt = 0; nt < 4; ++nt) {
      v16h a  = fragA(qs + h * 2048, mt * 16, 32, 0, lane);
      v16h bb = fragB(ks + h * 2048, nt * 16, 32, 0, lane);
      v8f acc = {};
      p[i][nt] = wmma_f16(a, bb, acc);
    }
    for (int nt = 0; nt < 4; ++nt) {
      int n = nt * 16 + laneN;
      for (int r = 0; r < 8; ++r) {
        int m = mt * 16 + r + laneHi * 8;
        float v = p[i][nt][r];
        if (n < NW) {
          if (m < NW)
            v += biasTbl[h * 2401 + m * 49 + n] +
                 mask[(size_t)win * 2401 + m * 49 + n];
        } else {
          v = -1e30f;                 // mask padded K columns
        }
        p[i][nt][r] = v;
      }
    }
    // row softmax: row lives in one 16-lane half; reduce with shfl_xor 1..8
    for (int r = 0; r < 8; ++r) {
      float mx = fmaxf(fmaxf(p[i][0][r], p[i][1][r]), fmaxf(p[i][2][r], p[i][3][r]));
      for (int d = 1; d < 16; d <<= 1) mx = fmaxf(mx, __shfl_xor(mx, d));
      float sum = 0.f;
      for (int nt = 0; nt < 4; ++nt) {
        float e = __expf(p[i][nt][r] - mx);
        p[i][nt][r] = e;
        sum += e;
      }
      for (int d = 1; d < 16; d <<= 1) sum += __shfl_xor(sum, d);
      float inv = 1.f / sum;
      for (int nt = 0; nt < 4; ++nt) p[i][nt][r] *= inv;
    }
  }
  __syncthreads();   // everyone done reading qs/ks -> safe to overwrite with as_

  // ---- 4. spill probs (f16) to as_ [h][m 64][n 64] ----
  for (int i = 0; i < 2; ++i) {
    int s = w + 8 * i; int h = s >> 2, mt = s & 3;
    for (int nt = 0; nt < 4; ++nt) {
      int n = nt * 16 + laneN;
      for (int r = 0; r < 8; ++r) {
        int m = mt * 16 + r + laneHi * 8;
        as_[h * 4096 + m * 64 + n] = (_Float16)p[i][nt][r];
      }
    }
  }
  __syncthreads();

  // ---- 5. AV: attn[64x64] @ v[64x32]; 32 C-tiles, 4 per wave ----
  for (int i = 0; i < 4; ++i) {
    int t = w + 8 * i;               // 0..31
    int h = t >> 3, mt = (t >> 1) & 3, dt = t & 1;
    v8f acc = {};
    for (int kt = 0; kt < 2; ++kt) {
      v16h a  = fragA(as_ + h * 4096, mt * 16, 64, kt * 32, lane);
      v16h bb = fragB(vts + h * 2048, dt * 16, 64, kt * 32, lane);
      acc = wmma_f16(a, bb, acc);
    }
    int col = h * 32 + dt * 16 + laneN;
    for (int r = 0; r < 8; ++r) {
      int m = mt * 16 + r + laneHi * 8;
      ys[m * 128 + col] = (_Float16)acc[r];
    }
  }
  __syncthreads();

  // ---- 6. output projection: [64x128] @ [128x128]; 32 C-tiles, 4 per wave ----
  for (int i = 0; i < 4; ++i) {
    int t = w + 8 * i;               // 0..31
    int mt = t & 3, nt = t >> 2;
    v8f acc = {};
    for (int kt = 0; kt < 4; ++kt) {
      v16h a  = fragA(ys, mt * 16, 128, kt * 32, lane);
      v16h bb = *(const v16h*)(wproj + (((size_t)nt * 4 + kt) * 32 + lane) * 16);
      acc = wmma_f16(a, bb, acc);
    }
    int n = nt * 16 + laneN;
    float pb = proj_b[n];
    for (int r = 0; r < 8; ++r) {
      int m = mt * 16 + r + laneHi * 8;
      if (m < NW) out[((size_t)b * NW + m) * CSZ + n] = acc[r] + pb;
    }
  }
}

extern "C" void kernel_launch(void* const* d_in, const int* in_sizes, int n_in,
                              void* d_out, int out_size, void* d_ws, size_t ws_size,
                              hipStream_t stream) {
  const float* inputs  = (const float*)d_in[0];
  const float* mask    = (const float*)d_in[1];
  const float* qkv_w   = (const float*)d_in[2];
  const float* qkv_b   = (const float*)d_in[3];
  const float* proj_w  = (const float*)d_in[4];
  const float* proj_b  = (const float*)d_in[5];
  const float* rel_tab = (const float*)d_in[6];
  float* out = (float*)d_out;

  char* ws = (char*)d_ws;
  _Float16* wqkv  = (_Float16*)ws;               // 98304 B
  _Float16* wproj = (_Float16*)(ws + 98304);     // 32768 B
  float*    biasT = (float*)(ws + 131072);       // 38416 B

  hipLaunchKernelGGL(swin_prep_kernel, dim3(128), dim3(256), 0, stream,
                     qkv_w, proj_w, rel_tab, wqkv, wproj, biasT);
  hipLaunchKernelGGL(swin_attn_kernel, dim3(2048), dim3(256), 0, stream,
                     inputs, mask, qkv_b, proj_b, wqkv, wproj, biasT, out);
}
